// FlashSelfAttention_19000935317632
// MI455X (gfx1250) — matrix operand order repaired and checked by reading, not compiled, not run on hardware
//
#include <hip/hip_runtime.h>

// ---------------------------------------------------------------------------
// Flash self-attention with ALiBi-sqrt causal bias for MI455X (gfx1250).
// Shapes fixed by the reference: B=2, S=2048, H=16, D=128, fp32 in/out.
// Strategy: bf16 WMMA (v_wmma_f32_16x16x32_bf16) flash attention.
//   block = 256 threads = 8 wave32; block owns (b, h, 128-row Q tile)
//   wave  = 16 Q rows; key loop in 64-key tiles staged via LDS.
// LDS pool (48KB) aliases the prologue Q buffer (32KB) with the per-tile
// K (16KB) + V^T (16KB) + P (16KB) buffers.
// Row reductions use v_permlane16_b32 (VALU) instead of ds_bpermute.
// ---------------------------------------------------------------------------

typedef __attribute__((ext_vector_type(16))) __bf16 v16bf;
typedef __attribute__((ext_vector_type(8)))  __bf16 v8bf;
typedef __attribute__((ext_vector_type(4)))  __bf16 v4bf;
typedef __attribute__((ext_vector_type(8)))  float  v8f;

#define B_      2
#define S_      2048
#define H_      16
#define D_      128
#define QTILE   128      // Q rows per block (16 per wave x 8 waves)
#define KTILE   64       // keys per iteration
#define NTHREADS 256

#define FP16MIN (-65504.0f)

// native f32 -> bf16 (RTNE) conversion (lowers to v_cvt_pk_bf16_f32)
static __device__ __forceinline__ __bf16 f2bf(float f) { return (__bf16)f; }

// ---- in-row (16-lane) butterfly shuffles on the VALU ----
#if __has_builtin(__builtin_amdgcn_permlane16)
static __device__ __forceinline__ float lane_xor(float x, unsigned s0, unsigned s1) {
  unsigned u = __builtin_bit_cast(unsigned, x);
  unsigned r = __builtin_amdgcn_permlane16(u, u, s0, s1, false, false);
  return __builtin_bit_cast(float, r);
}
static __device__ __forceinline__ float rowmax16(float x) {
  x = fmaxf(x, lane_xor(x, 0x67452301u, 0xEFCDAB89u));  // xor 1
  x = fmaxf(x, lane_xor(x, 0x54761032u, 0xDCFE98BAu));  // xor 2
  x = fmaxf(x, lane_xor(x, 0x32107654u, 0xBA98FEDCu));  // xor 4
  x = fmaxf(x, lane_xor(x, 0xFEDCBA98u, 0x76543210u));  // xor 8
  return x;
}
static __device__ __forceinline__ float rowsum16(float x) {
  x += lane_xor(x, 0x67452301u, 0xEFCDAB89u);
  x += lane_xor(x, 0x54761032u, 0xDCFE98BAu);
  x += lane_xor(x, 0x32107654u, 0xBA98FEDCu);
  x += lane_xor(x, 0xFEDCBA98u, 0x76543210u);
  return x;
}
#else
static __device__ __forceinline__ float rowmax16(float x) {
  for (int off = 1; off < 16; off <<= 1) x = fmaxf(x, __shfl_xor(x, off, 16));
  return x;
}
static __device__ __forceinline__ float rowsum16(float x) {
  for (int off = 1; off < 16; off <<= 1) x += __shfl_xor(x, off, 16);
  return x;
}
#endif

// Assemble a 16-element bf16 fragment from two 8-element LDS chunks.
static __device__ __forceinline__ v16bf ld2(const __bf16* p0, const __bf16* p1) {
  v8bf a = *(const v8bf*)p0;
  v8bf b = *(const v8bf*)p1;
  v16bf r;
#pragma unroll
  for (int i = 0; i < 8; ++i) { r[i] = a[i]; r[i + 8] = b[i]; }
  return r;
}

__global__ __launch_bounds__(NTHREADS)
void flash_attn_alibi_kernel(const float* __restrict__ Q,
                             const float* __restrict__ K,
                             const float* __restrict__ V,
                             float* __restrict__ Out) {
  // 48KB bf16 pool, aliased:
  //   prologue: sQ = pool[0 .. 16384)                  (128x128)
  //   loop:     sK = pool[0 .. 8192)                   (64 keys x 128 d)
  //             sVt= pool[8192 .. 16384)               (128 d x 64 keys)
  //             sP = pool[16384 .. 24576)              (8 waves x 16 x 64)
  __shared__ alignas(16) __bf16 pool[24576];
  __bf16* const sQ  = pool;
  __bf16* const sK  = pool;
  __bf16* const sVt = pool + KTILE * D_;
  __bf16* const sP  = pool + 2 * KTILE * D_;

  const int tid    = threadIdx.x;
  const int wave   = tid >> 5;
  const int lane   = tid & 31;
  const int lane16 = lane & 15;
  const int hg     = lane >> 4;          // half-wave group (0/1)

  const int bb = blockIdx.x;             // (b*H + h)*QT + qt
  const int QT = S_ / QTILE;
  const int qt = bb % QT;
  const int h  = (bb / QT) % H_;
  const int b  = bb / (QT * H_);

  const int qbase = qt * QTILE;          // block's first query row
  const int qw    = qbase + wave * 16;   // wave's first query row
  const int qwmax = qw + 15;

  const float scale = 0.08838834764831845f;             // 1/sqrt(128)
  const float slope = exp2f(-0.5f * (float)(h + 1));    // ALiBi slope, H=16

  // ---- stage Q tile (fp32 -> bf16), cooperative across all 8 waves ----
#pragma unroll 4
  for (int i = tid; i < QTILE * D_ / 4; i += NTHREADS) {
    const int m  = i >> 5;                               // D/4 = 32
    const int d4 = i & 31;
    const float4 qv = *(const float4*)(Q + ((size_t)((b * S_ + qbase + m) * H_ + h)) * D_ + d4 * 4);
    v4bf pk; pk[0] = f2bf(qv.x); pk[1] = f2bf(qv.y); pk[2] = f2bf(qv.z); pk[3] = f2bf(qv.w);
    *(v4bf*)&sQ[m * D_ + d4 * 4] = pk;
  }
  __syncthreads();

  // ---- load this wave's Q A-fragments (16 rows x 128 d as 4x [16x32]) ----
  v16bf qf[4];
#pragma unroll
  for (int c = 0; c < 4; ++c) {
    const __bf16* p = &sQ[(wave * 16 + lane16) * D_ + c * 32 + hg * 8];
    qf[c] = ld2(p, p + 16);
  }
  // NOTE: sQ is dead from here on; the loop's leading __syncthreads orders
  // the first overwrite of the pool after every wave has its Q fragments.

  // ---- online softmax state ----
  float m_i[8], l_i[8];
  v8f o[8] = {};                                         // 16 rows x 128 d accum
#pragma unroll
  for (int r = 0; r < 8; ++r) { m_i[r] = -1.0e30f; l_i[r] = 0.0f; }

  const int ktiles = 2 * (qt + 1);                       // causal: up to block diagonal

  for (int kt = 0; kt < ktiles; ++kt) {
    const int kb = kt * KTILE;
    __syncthreads();                                     // prev tile's reads done

    // prefetch next K tile (global_prefetch_b8 path); 64 rows x 512B
    if (kt + 1 < ktiles) {
      const float* pf = K + ((size_t)((b * S_ + kb + KTILE + (tid >> 2)) * H_ + h)) * D_ + (tid & 3) * 32;
      __builtin_prefetch(pf, 0, 1);
    }

    // ---- stage K tile (row-major) and V tile (transposed) ----
#pragma unroll 4
    for (int i = tid; i < KTILE * D_ / 4; i += NTHREADS) {
      const int key = i >> 5;
      const int d4  = i & 31;
      const size_t g = ((size_t)((b * S_ + kb + key) * H_ + h)) * D_ + d4 * 4;
      const float4 kv = *(const float4*)(K + g);
      v4bf pk; pk[0] = f2bf(kv.x); pk[1] = f2bf(kv.y); pk[2] = f2bf(kv.z); pk[3] = f2bf(kv.w);
      *(v4bf*)&sK[key * D_ + d4 * 4] = pk;
      const float4 vv = *(const float4*)(V + g);
      sVt[(d4 * 4 + 0) * KTILE + key] = f2bf(vv.x);
      sVt[(d4 * 4 + 1) * KTILE + key] = f2bf(vv.y);
      sVt[(d4 * 4 + 2) * KTILE + key] = f2bf(vv.z);
      sVt[(d4 * 4 + 3) * KTILE + key] = f2bf(vv.w);
    }
    __syncthreads();

    if (kb > qwmax) continue;                            // wave-uniform causal skip

    // ---- S = Q K^T : four 16x16 C tiles, 4-deep WMMA chain over D ----
    v8f s[4];
#pragma unroll
    for (int t = 0; t < 4; ++t) {
      v8f acc = {};
#pragma unroll
      for (int c = 0; c < 4; ++c) {
        const __bf16* kp = &sK[(t * 16 + lane16) * D_ + c * 32 + hg * 8];
        v16bf kf = ld2(kp, kp + 16);
        acc = __builtin_amdgcn_wmma_f32_16x16x32_bf16(
            false, qf[c], false, kf, (short)0, acc, false, false);
      }
      s[t] = acc;
    }

    // ---- online softmax (row lives in a 16-lane half-wave) ----
    float alpha[8];
#pragma unroll
    for (int r = 0; r < 8; ++r) {
      const int qrow = qw + r + 8 * hg;
      float x[4];
#pragma unroll
      for (int t = 0; t < 4; ++t) {
        const int kc = kb + t * 16 + lane16;
        const float bias = (kc <= qrow) ? (-slope * sqrtf((float)(qrow - kc))) : FP16MIN;
        x[t] = s[t][r] * scale + bias;
      }
      const float rmax = rowmax16(fmaxf(fmaxf(x[0], x[1]), fmaxf(x[2], x[3])));

      const float mnew = fmaxf(m_i[r], rmax);
      const float a    = __expf(m_i[r] - mnew);
      float p[4], psum = 0.0f;
#pragma unroll
      for (int t = 0; t < 4; ++t) { p[t] = __expf(x[t] - mnew); psum += p[t]; }
      const float rsum = rowsum16(psum);

      l_i[r] = l_i[r] * a + rsum;
      m_i[r] = mnew;
      alpha[r] = a;

      // stash probs (C layout -> row-major LDS); same-wave DS ops are in-order
      __bf16* pp = &sP[wave * (16 * KTILE) + (r + 8 * hg) * KTILE + lane16];
      pp[0]  = f2bf(p[0]);
      pp[16] = f2bf(p[1]);
      pp[32] = f2bf(p[2]);
      pp[48] = f2bf(p[3]);
    }

    // ---- O = diag(alpha) O + P V : two 16x32 P fragments over 64 keys ----
    const __bf16* pb = &sP[wave * (16 * KTILE) + lane16 * KTILE + hg * 8];
    const v16bf pf0 = ld2(pb,      pb + 16);             // keys 0..31 of tile
    const v16bf pf1 = ld2(pb + 32, pb + 48);             // keys 32..63 of tile
#pragma unroll
    for (int nb = 0; nb < 8; ++nb) {
#pragma unroll
      for (int r = 0; r < 8; ++r) o[nb][r] *= alpha[r];
      const __bf16* vp = &sVt[(nb * 16 + lane16) * KTILE + hg * 8];
      v16bf vf0 = ld2(vp,      vp + 16);                 // keys 0..31
      v16bf vf1 = ld2(vp + 32, vp + 48);                 // keys 32..63
      o[nb] = __builtin_amdgcn_wmma_f32_16x16x32_bf16(
          false, pf0, false, vf0, (short)0, o[nb], false, false);
      o[nb] = __builtin_amdgcn_wmma_f32_16x16x32_bf16(
          false, pf1, false, vf1, (short)0, o[nb], false, false);
    }
  }

  // ---- normalize and write out: out[b, q, h, d] ----
#pragma unroll
  for (int r = 0; r < 8; ++r) {
    const int qrow = qw + r + 8 * hg;
    const float inv = 1.0f / l_i[r];
    const size_t base = ((size_t)((b * S_ + qrow) * H_ + h)) * D_;
#pragma unroll
    for (int nb = 0; nb < 8; ++nb)
      Out[base + nb * 16 + lane16] = o[nb][r] * inv;
  }
}

extern "C" void kernel_launch(void* const* d_in, const int* in_sizes, int n_in,
                              void* d_out, int out_size, void* d_ws, size_t ws_size,
                              hipStream_t stream) {
  (void)in_sizes; (void)n_in; (void)d_ws; (void)ws_size; (void)out_size;
  const float* q = (const float*)d_in[0];
  const float* k = (const float*)d_in[1];
  const float* v = (const float*)d_in[2];
  float* out = (float*)d_out;

  dim3 grid(B_ * H_ * (S_ / QTILE));   // 512 blocks
  dim3 block(NTHREADS);                // 8 wave32
  hipLaunchKernelGGL(flash_attn_alibi_kernel, grid, block, 0, stream, q, k, v, out);
}
